// Mamba2_53541062312498
// MI455X (gfx1250) — 8-bit matrix operand order repaired and checked by reading, compiled, not run on hardware
//
#include <hip/hip_runtime.h>
#include <hip/hip_bf16.h>


#define B_       64
#define DMODEL   2048
#define DSSM     4096
#define NHEADS_  64
#define HEADDIM_ 64
#define DSTATE_  128
#define DINPROJ  8512

typedef __attribute__((ext_vector_type(8))) int          v8i;
typedef __attribute__((ext_vector_type(4))) int          v4i_;
typedef __attribute__((ext_vector_type(2))) int          v2i_;
typedef __attribute__((ext_vector_type(4))) unsigned int u32x4;
typedef __attribute__((ext_vector_type(8))) int          i32x8;
typedef __attribute__((ext_vector_type(4))) int          i32x4;

#if __has_builtin(__builtin_amdgcn_tensor_load_to_lds) && \
    __has_builtin(__builtin_amdgcn_s_wait_tensorcnt)
#define HAVE_TDM 1
#else
#define HAVE_TDM 0
#endif

__device__ __forceinline__ float mkscale(float absmax) {
    float s = absmax * (1.0f / 127.0f);
    return (s == 0.0f) ? 1.0f : s;
}
__device__ __forceinline__ float fqv(float v, float scale) {
    return rintf(v / scale) * scale;
}

__global__ __launch_bounds__(256)
void rowquant_kernel(const float* __restrict__ W, int K,
                     signed char* __restrict__ Wq, float* __restrict__ scales) {
    int row = blockIdx.x;
    const float* wr = W + (size_t)row * K;
    float m = 0.0f;
    for (int i = threadIdx.x; i < K; i += 256)
        m = fmaxf(m, fabsf(__builtin_nontemporal_load(wr + i)));
    __shared__ float red[256];
    red[threadIdx.x] = m;
    __syncthreads();
    for (int s = 128; s > 0; s >>= 1) {
        if (threadIdx.x < s) red[threadIdx.x] = fmaxf(red[threadIdx.x], red[threadIdx.x + s]);
        __syncthreads();
    }
    float scale = mkscale(red[0]);
    if (threadIdx.x == 0) scales[row] = scale;
    float inv = 1.0f / scale;
    signed char* qr = Wq + (size_t)row * K;
    for (int i = threadIdx.x; i < K; i += 256)
        qr[i] = (signed char)(int)rintf(__builtin_nontemporal_load(wr + i) * inv);
}

__global__ __launch_bounds__(256)
void absrow_kernel(const float* __restrict__ src, int rowStride, int off, int len,
                   float* __restrict__ out) {
    int row = blockIdx.x;
    const float* p = src + (size_t)row * rowStride + off;
    float m = 0.0f;
    for (int i = threadIdx.x; i < len; i += 256) m = fmaxf(m, fabsf(p[i]));
    __shared__ float red[256];
    red[threadIdx.x] = m;
    __syncthreads();
    for (int s = 128; s > 0; s >>= 1) {
        if (threadIdx.x < s) red[threadIdx.x] = fmaxf(red[threadIdx.x], red[threadIdx.x + s]);
        __syncthreads();
    }
    if (threadIdx.x == 0) out[row] = red[0];
}

__device__ __forceinline__ v8i load_bfrag(const signed char* bp) {
    v4i_ b0 = __builtin_nontemporal_load((const v4i_*)(bp));
    v4i_ b1 = __builtin_nontemporal_load((const v4i_*)(bp + 32));
    v8i r = {b0.x, b0.y, b0.z, b0.w, b1.x, b1.y, b1.z, b1.w};
    return r;
}
__device__ __forceinline__ v8i load_afrag(const signed char* ap) {
    v4i_ a0 = *(const v4i_*)(ap);
    v4i_ a1 = *(const v4i_*)(ap + 32);
    v8i r = {a0.x, a0.y, a0.z, a0.w, a1.x, a1.y, a1.z, a1.w};
    return r;
}

__global__ __launch_bounds__(256)
void gemm_q8_kernel(const signed char* __restrict__ Aq, const signed char* __restrict__ Wq,
                    const float* __restrict__ sa, const float* __restrict__ sw,
                    float* __restrict__ out, int N, int K) {
    int wave  = (blockIdx.x * blockDim.x + threadIdx.x) >> 5;
    int lane  = threadIdx.x & 31;
    int ntile = wave;
    if (ntile * 16 >= N) return;

    int col = ntile * 16 + (lane & 15);
    int hs  = lane >> 4;
    const signed char* bp  = Wq + (size_t)col * K + hs * 16;
    const signed char* ap0 = Aq + (size_t)(lane & 15) * K + hs * 16;
    const size_t mstep = (size_t)16 * K;

    v8i acc[4];
#pragma unroll
    for (int m = 0; m < 4; ++m) acc[m] = (v8i)(0);

    v8i bf0 = load_bfrag(bp);
    v8i a00 = load_afrag(ap0);
    v8i a01 = load_afrag(ap0 + mstep);
    v8i a02 = load_afrag(ap0 + 2 * mstep);
    v8i a03 = load_afrag(ap0 + 3 * mstep);

    for (int k = 0; k < K; k += 128) {
        int k1 = k + 64;
        __builtin_prefetch(bp + k + 2048, 0, 1);
        v8i bf1 = load_bfrag(bp + k1);
        v8i a10 = load_afrag(ap0 + k1);
        v8i a11 = load_afrag(ap0 + mstep + k1);
        v8i a12 = load_afrag(ap0 + 2 * mstep + k1);
        v8i a13 = load_afrag(ap0 + 3 * mstep + k1);

        acc[0] = __builtin_amdgcn_wmma_i32_16x16x64_iu8(true, a00, true, bf0, acc[0], false, false);
        acc[1] = __builtin_amdgcn_wmma_i32_16x16x64_iu8(true, a01, true, bf0, acc[1], false, false);
        acc[2] = __builtin_amdgcn_wmma_i32_16x16x64_iu8(true, a02, true, bf0, acc[2], false, false);
        acc[3] = __builtin_amdgcn_wmma_i32_16x16x64_iu8(true, a03, true, bf0, acc[3], false, false);

        int k2 = (k + 128 < K) ? (k + 128) : 0;
        bf0 = load_bfrag(bp + k2);
        a00 = load_afrag(ap0 + k2);
        a01 = load_afrag(ap0 + mstep + k2);
        a02 = load_afrag(ap0 + 2 * mstep + k2);
        a03 = load_afrag(ap0 + 3 * mstep + k2);

        acc[0] = __builtin_amdgcn_wmma_i32_16x16x64_iu8(true, a10, true, bf1, acc[0], false, false);
        acc[1] = __builtin_amdgcn_wmma_i32_16x16x64_iu8(true, a11, true, bf1, acc[1], false, false);
        acc[2] = __builtin_amdgcn_wmma_i32_16x16x64_iu8(true, a12, true, bf1, acc[2], false, false);
        acc[3] = __builtin_amdgcn_wmma_i32_16x16x64_iu8(true, a13, true, bf1, acc[3], false, false);
    }

    float swn = sw[col];
#pragma unroll
    for (int m = 0; m < 4; ++m) {
#pragma unroll
        for (int r = 0; r < 8; ++r) {
            int row = m * 16 + r + hs * 8;
            out[(size_t)row * N + col] = (float)acc[m][r] * sa[row] * swn;
        }
    }
}

__global__ __launch_bounds__(128)
void dt_kernel(const float* __restrict__ zx, const float* __restrict__ sxbc_abs,
               const float* __restrict__ dt_bias, const float* __restrict__ A_log,
               float* __restrict__ dtv, float* __restrict__ scl, float* __restrict__ BCv) {
    int b = blockIdx.x;
    int t = threadIdx.x;
    float sxbc = mkscale(sxbc_abs[b]);
    float bq = fqv(zx[(size_t)b * DINPROJ + 2 * DSSM        + t], sxbc);
    float cq = fqv(zx[(size_t)b * DINPROJ + 2 * DSSM + 128  + t], sxbc);
    __shared__ float red[128];
    red[t] = bq * cq;
    __syncthreads();
    for (int s = 64; s > 0; s >>= 1) {
        if (t < s) red[t] += red[t + s];
        __syncthreads();
    }
    if (t == 0) BCv[b] = red[0];
    if (t < NHEADS_) {
        int h = t;
        float dt = zx[(size_t)b * DINPROJ + (DINPROJ - NHEADS_) + h] + dt_bias[h];
        float sp;
        if (dt < -2.0f)     sp = 0.0f;
        else if (dt > 2.0f) sp = dt;
        else sp = 0.69314718055994531f + 0.5f * dt + dt * dt * 0.125f
                  + dt * dt * dt * (1.0f / 48.0f);
        float A  = -__expf(A_log[h]);
        float dA = fmaxf(sp * A, -10000.0f);
        float e = 1.0f + dA * (1.0f + dA * (0.5f + dA * ((1.0f / 6.0f)
                  + dA * ((1.0f / 24.0f) + dA * (1.0f / 120.0f)))));
        scl[b * NHEADS_ + h] = fminf(fmaxf(e, 0.0f), 1.0f);
        dtv[b * NHEADS_ + h] = sp;
    }
}

__global__ __launch_bounds__(128)
void ssm_y_kernel(const float* __restrict__ state, const float* __restrict__ zx,
                  const float* __restrict__ sxbc_abs, const float* __restrict__ dtv,
                  const float* __restrict__ scl, const float* __restrict__ BCv,
                  float* __restrict__ t1, unsigned int* __restrict__ abs1) {
    int bh = blockIdx.x;
    int b  = bh >> 6;
    int t  = threadIdx.x;
    int w  = t >> 5, lane = t & 31;
    const float* st = state + (size_t)bh * (HEADDIM_ * DSTATE_);
    float sxbc = mkscale(sxbc_abs[b]);
    float cq = fqv(zx[(size_t)b * DINPROJ + 2 * DSSM + 128 + t], sxbc);

#if HAVE_TDM
    __shared__ float sbuf[HEADDIM_ * DSTATE_];
    if (threadIdx.x < 32) {
        unsigned long long ga = (unsigned long long)(uintptr_t)st;
        u32x4 g0;
        g0.x = 1u;
        g0.y = (unsigned int)(uintptr_t)&sbuf[0];
        g0.z = (unsigned int)ga;
        g0.w = (unsigned int)((ga >> 32) & 0x01FFFFFFu) | (2u << 30);
        i32x8 g1;
        g1.s0 = (int)(2u << 16);
        g1.s1 = (int)(DSTATE_ << 16);
        g1.s2 = (int)(HEADDIM_ << 16);
        g1.s3 = (int)(DSTATE_ << 16);
        g1.s4 = HEADDIM_;
        g1.s5 = DSTATE_;
        g1.s6 = 0;
        g1.s7 = 0;
        i32x4 z4 = {0, 0, 0, 0};
        i32x8 z8 = {0, 0, 0, 0, 0, 0, 0, 0};
        __builtin_amdgcn_tensor_load_to_lds(g0, g1, z4, z4, z8, 0);
        __builtin_amdgcn_s_wait_tensorcnt(0);
    }
    __syncthreads();
#endif

    __shared__ float part[HEADDIM_][4];
    for (int p = 0; p < HEADDIM_; ++p) {
#if HAVE_TDM
        float v = sbuf[p * DSTATE_ + t] * cq;
#else
        float v = __builtin_nontemporal_load(st + p * DSTATE_ + t) * cq;
#endif
        v += __shfl_xor(v, 16, 32);
        v += __shfl_xor(v,  8, 32);
        v += __shfl_xor(v,  4, 32);
        v += __shfl_xor(v,  2, 32);
        v += __shfl_xor(v,  1, 32);
        if (lane == 0) part[p][w] = v;
    }
    __syncthreads();
    if (t < HEADDIM_) {
        int p = t, h = bh & 63;
        float dot = part[p][0] + part[p][1] + part[p][2] + part[p][3];
        float xq  = fqv(zx[(size_t)b * DINPROJ + DSSM + h * HEADDIM_ + p], sxbc);
        float y   = scl[bh] * dot + dtv[bh] * xq * BCv[b];
        t1[(size_t)b * DSSM + h * HEADDIM_ + p] = y;
        atomicMax(abs1 + b, __float_as_uint(fabsf(y)));
    }
}

__global__ __launch_bounds__(256)
void ymid_kernel(const float* __restrict__ t1, const unsigned int* __restrict__ abs1,
                 const float* __restrict__ zx, const float* __restrict__ sxbc_abs,
                 const float* __restrict__ Dv,
                 signed char* __restrict__ yq, float* __restrict__ sy) {
    int b = blockIdx.x;
    float s1   = mkscale(__uint_as_float(abs1[b]));
    float sxbc = mkscale(sxbc_abs[b]);
    float vals[DSSM / 256];
    float m = 0.0f;
#pragma unroll
    for (int j = 0; j < DSSM / 256; ++j) {
        int i = j * 256 + threadIdx.x;
        float y1 = fqv(t1[(size_t)b * DSSM + i], s1);
        float xq = fqv(zx[(size_t)b * DINPROJ + DSSM + i], sxbc);
        float z  = zx[(size_t)b * DINPROJ + i];
        float v  = (y1 + Dv[i] * xq) * fmaxf(z, 0.0f);
        vals[j] = v;
        m = fmaxf(m, fabsf(v));
    }
    __shared__ float red[256];
    red[threadIdx.x] = m;
    __syncthreads();
    for (int s = 128; s > 0; s >>= 1) {
        if (threadIdx.x < s) red[threadIdx.x] = fmaxf(red[threadIdx.x], red[threadIdx.x + s]);
        __syncthreads();
    }
    float scale = mkscale(red[0]);
    if (threadIdx.x == 0) sy[b] = scale;
    float inv = 1.0f / scale;
#pragma unroll
    for (int j = 0; j < DSSM / 256; ++j) {
        int i = j * 256 + threadIdx.x;
        yq[(size_t)b * DSSM + i] = (signed char)(int)rintf(vals[j] * inv);
    }
}

extern "C" void kernel_launch(void* const* d_in, const int* in_sizes, int n_in,
                              void* d_out, int out_size, void* d_ws, size_t ws_size,
                              hipStream_t stream) {
    (void)in_sizes; (void)n_in; (void)out_size; (void)ws_size;
    const float* hidden  = (const float*)d_in[0];
    const float* state   = (const float*)d_in[1];
    const float* W_in    = (const float*)d_in[2];
    const float* dt_bias = (const float*)d_in[3];
    const float* A_log   = (const float*)d_in[4];
    const float* Dv      = (const float*)d_in[5];
    const float* W_out   = (const float*)d_in[6];
    float* out = (float*)d_out;

    char* ws = (char*)d_ws;
    size_t o = 0;
    auto take = [&](size_t bytes) { size_t r = o; o += (bytes + 255) & ~(size_t)255; return r; };
    signed char* whq  = (signed char*)(ws + take((size_t)DINPROJ * DMODEL));
    signed char* woq  = (signed char*)(ws + take((size_t)DMODEL * DSSM));
    signed char* hq   = (signed char*)(ws + take((size_t)B_ * DMODEL));
    signed char* yq   = (signed char*)(ws + take((size_t)B_ * DSSM));
    float* zx   = (float*)(ws + take((size_t)B_ * DINPROJ * 4));
    float* t1   = (float*)(ws + take((size_t)B_ * DSSM * 4));
    float* sw_in  = (float*)(ws + take(DINPROJ * 4));
    float* sw_out = (float*)(ws + take(DMODEL * 4));
    float* sh     = (float*)(ws + take(B_ * 4));
    float* sy     = (float*)(ws + take(B_ * 4));
    float* sxbc   = (float*)(ws + take(B_ * 4));
    unsigned int* abs1 = (unsigned int*)(ws + take(B_ * 4));
    float* dtv    = (float*)(ws + take(B_ * NHEADS_ * 4));
    float* scl    = (float*)(ws + take(B_ * NHEADS_ * 4));
    float* BCv    = (float*)(ws + take(B_ * 4));

    (void)hipMemsetAsync(abs1, 0, B_ * 4, stream);

    rowquant_kernel<<<B_,      256, 0, stream>>>(hidden, DMODEL, hq,  sh);
    rowquant_kernel<<<DINPROJ, 256, 0, stream>>>(W_in,   DMODEL, whq, sw_in);
    rowquant_kernel<<<DMODEL,  256, 0, stream>>>(W_out,  DSSM,   woq, sw_out);

    {
        int ntiles = DINPROJ / 16;
        int blocks = (ntiles + 7) / 8;
        gemm_q8_kernel<<<blocks, 256, 0, stream>>>(hq, whq, sh, sw_in, zx, DINPROJ, DMODEL);
    }

    absrow_kernel<<<B_, 256, 0, stream>>>(zx, DINPROJ, DSSM, DSSM + 2 * DSTATE_, sxbc);

    dt_kernel<<<B_, 128, 0, stream>>>(zx, sxbc, dt_bias, A_log, dtv, scl, BCv);

    ssm_y_kernel<<<B_ * NHEADS_, 128, 0, stream>>>(state, zx, sxbc, dtv, scl, BCv, t1, abs1);

    ymid_kernel<<<B_, 256, 0, stream>>>(t1, abs1, zx, sxbc, Dv, yq, sy);

    gemm_q8_kernel<<<(DMODEL / 16) / 8, 256, 0, stream>>>(yq, woq, sy, sw_out, out, DMODEL, DSSM);
}
